// Attend_30030411333687
// MI455X (gfx1250) — matrix operand order repaired and checked by reading, not compiled, not run on hardware
//
#include <hip/hip_runtime.h>

// CDNA5 / gfx1250 fused attention:
//   out = softmax(QK^T * scale + prev_attn + bias, mask & causal) @ V
// B=1, H=16, N=2048, D=64, f32 in/out. f16 WMMA for both matmuls,
// flash-style online softmax, causal j-block skipping, NT loads for the
// read-once [H,N,N] tensors.

typedef __attribute__((ext_vector_type(16))) _Float16 v16h;
typedef __attribute__((ext_vector_type(8)))  float    v8f;

#define SEQ_N 2048
#define DHEAD 64
#define MASKV (-3.402823466e38f)

__global__ __launch_bounds__(128) void attend_fused(
    const float* __restrict__ Q, const float* __restrict__ K,
    const float* __restrict__ V, const unsigned char* __restrict__ Msk,
    const float* __restrict__ Bias, const float* __restrict__ Prev,
    float* __restrict__ Out)
{
  // per-wave 16x32 f16 staging tile for the P (probabilities) transpose
  __shared__ _Float16 plds[4][16 * 32];

  const int lane = threadIdx.x & 31;
  const int wid  = threadIdx.x >> 5;
  const int gw   = blockIdx.x * 4 + wid;   // 0..2047 global wave id
  const int h    = gw >> 7;                // head
  const int q0   = (gw & 127) << 4;        // first query row of this tile
  const int nn   = lane & 15;
  const int hh   = lane >> 4;              // lane half select
  _Float16* pl = &plds[wid][0];

  // ---- Q A-fragments (16x32 f16 WMMA layout), two 32-wide d-chunks ----
  v16h aq0, aq1;
  {
    const float* qrow = Q + ((size_t)h * SEQ_N + q0 + nn) * DHEAD;  // M = nn
    #pragma unroll
    for (int e = 0; e < 16; ++e) {
      const int kk = (e < 8) ? (8 * hh + e) : (8 + 8 * hh + e);
      aq0[e] = (_Float16)qrow[kk];
      aq1[e] = (_Float16)qrow[32 + kk];
    }
  }

  v8f o0 = {}, o1 = {}, o2 = {}, o3 = {};   // O accumulator, 16x64 f32
  float m[8], ls[8];                        // per-row running max / sum
  #pragma unroll
  for (int r = 0; r < 8; ++r) { m[r] = MASKV; ls[r] = 0.0f; }

  const int njb = (q0 >> 5) + 1;            // causal: only j0 <= q0+15 needed
  for (int jb = 0; jb < njb; ++jb) {
    const int j0 = jb << 5;

    // ---- S = Q K^T for a 16x32 block: two 16x16 tiles, chained over d ----
    v8f s0 = {}, s1 = {};
    {
      const float* kr0 = K + ((size_t)h * SEQ_N + j0 + nn) * DHEAD;  // N = nn
      const float* kr1 = kr0 + 16 * DHEAD;
      v16h b00, b01, b10, b11;
      #pragma unroll
      for (int e = 0; e < 16; ++e) {
        const int kk = (e < 8) ? (8 * hh + e) : (8 + 8 * hh + e);  // B layout: table K order
        b00[e] = (_Float16)kr0[kk];
        b01[e] = (_Float16)kr0[32 + kk];
        b10[e] = (_Float16)kr1[kk];
        b11[e] = (_Float16)kr1[32 + kk];
      }
      s0 = __builtin_amdgcn_wmma_f32_16x16x32_f16(false, aq0, false, b00, (short)0, s0, false, false);
      s0 = __builtin_amdgcn_wmma_f32_16x16x32_f16(false, aq1, false, b01, (short)0, s0, false, false);
      s1 = __builtin_amdgcn_wmma_f32_16x16x32_f16(false, aq0, false, b10, (short)0, s1, false, false);
      s1 = __builtin_amdgcn_wmma_f32_16x16x32_f16(false, aq1, false, b11, (short)0, s1, false, false);
    }

    // ---- logits + masks + online softmax (row = r + 8*hh in C layout) ----
    #pragma unroll
    for (int r = 0; r < 8; ++r) {
      const int i = q0 + r + 8 * hh;
      const size_t rb = ((size_t)h * SEQ_N + i) * SEQ_N;
      const size_t mb = (size_t)i * SEQ_N;
      const int jA = j0 + nn, jB = j0 + 16 + nn;
      float t0 = s0[r] * 0.125f + __builtin_nontemporal_load(Prev + rb + jA)
                                + __builtin_nontemporal_load(Bias + rb + jA);
      float t1 = s1[r] * 0.125f + __builtin_nontemporal_load(Prev + rb + jB)
                                + __builtin_nontemporal_load(Bias + rb + jB);
      if (!Msk[mb + jA] || jA > i) t0 = MASKV;
      if (!Msk[mb + jB] || jB > i) t1 = MASKV;

      float mx = fmaxf(t0, t1);
      #pragma unroll
      for (int d = 1; d < 16; d <<= 1) mx = fmaxf(mx, __shfl_xor(mx, d, 32));
      const float mn    = fmaxf(m[r], mx);
      const float alpha = __expf(m[r] - mn);
      m[r] = mn;
      const float p0 = __expf(t0 - mn);
      const float p1 = __expf(t1 - mn);
      float rs = p0 + p1;
      #pragma unroll
      for (int d = 1; d < 16; d <<= 1) rs += __shfl_xor(rs, d, 32);
      ls[r] = ls[r] * alpha + rs;
      o0[r] *= alpha; o1[r] *= alpha; o2[r] *= alpha; o3[r] *= alpha;

      pl[(r + 8 * hh) * 32 + nn]      = (_Float16)p0;   // stage P for transpose
      pl[(r + 8 * hh) * 32 + 16 + nn] = (_Float16)p1;
    }
    asm volatile("s_wait_dscnt 0" ::: "memory");

    // ---- P (A-fragment) from LDS; V B-fragments; O += P V ----
    v16h pa;
    #pragma unroll
    for (int e = 0; e < 16; ++e) {
      const int kk = 16 * hh + e;
      pa[e] = pl[nn * 32 + kk];                 // row M = nn
    }
    const float* vbase = V + ((size_t)h * SEQ_N + j0 + 16 * hh) * DHEAD + nn;
    v16h w0, w1, w2, w3;
    #pragma unroll
    for (int e = 0; e < 16; ++e) {
      const float* vr = vbase + (size_t)e * DHEAD;   // key row j0 + 16*hh + e
      w0[e] = (_Float16)vr[0];
      w1[e] = (_Float16)vr[16];
      w2[e] = (_Float16)vr[32];
      w3[e] = (_Float16)vr[48];
    }
    o0 = __builtin_amdgcn_wmma_f32_16x16x32_f16(false, pa, false, w0, (short)0, o0, false, false);
    o1 = __builtin_amdgcn_wmma_f32_16x16x32_f16(false, pa, false, w1, (short)0, o1, false, false);
    o2 = __builtin_amdgcn_wmma_f32_16x16x32_f16(false, pa, false, w2, (short)0, o2, false, false);
    o3 = __builtin_amdgcn_wmma_f32_16x16x32_f16(false, pa, false, w3, (short)0, o3, false, false);
    asm volatile("s_wait_dscnt 0" ::: "memory");  // keep LDS reuse ordered
  }

  // ---- normalize and write out[B,H,N,D] ----
  #pragma unroll
  for (int r = 0; r < 8; ++r) {
    const float inv = 1.0f / ls[r];
    const int i = q0 + r + 8 * hh;
    float* orow = Out + ((size_t)h * SEQ_N + i) * DHEAD;
    orow[nn]      = o0[r] * inv;
    orow[16 + nn] = o1[r] * inv;
    orow[32 + nn] = o2[r] * inv;
    orow[48 + nn] = o3[r] * inv;
  }
}

extern "C" void kernel_launch(void* const* d_in, const int* in_sizes, int n_in,
                              void* d_out, int out_size, void* d_ws, size_t ws_size,
                              hipStream_t stream) {
  (void)in_sizes; (void)n_in; (void)out_size; (void)d_ws; (void)ws_size;
  const float*         q    = (const float*)d_in[0];
  const float*         k    = (const float*)d_in[1];
  const float*         v    = (const float*)d_in[2];
  const unsigned char* msk  = (const unsigned char*)d_in[3];
  const float*         bias = (const float*)d_in[4];
  const float*         prev = (const float*)d_in[5];
  float*               out  = (float*)d_out;

  dim3 grid(512);   // 2048 waves total, one 16-row query tile per wave
  dim3 block(128);  // 4 waves per workgroup
  hipLaunchKernelGGL(attend_fused, grid, block, 0, stream,
                     q, k, v, msk, bias, prev, out);
}
